// DeltaLoss_19224273617653
// MI455X (gfx1250) — matrix operand order repaired and checked by reading, not compiled, not run on hardware
//
#include <hip/hip_runtime.h>

// ---------------- problem constants ----------------
#define NPTS   96
#define DDIM   512
#define NPAIR  9120              // 96*95
#define PPAD   9216              // 72*128 = 144*64
#define RB     128               // rows per block in LSE pass (8 waves x 16)
#define NCT    144               // column tiles of 64  (PPAD/64)
#define CSPLIT 12                // grid.y column splits
#define TPS    12                // tiles per split (12*12 = 144)
#define INV_TEMP 100.0f
#define EPSN   1e-8f

// LDS layout for the LSE pass (dynamic shared memory)
#define A_PITCH  1040u                    // 512 halfs + 16B pad  -> conflict-free frags
#define B_PITCH  1040u                    // TDM pad: 256 dwords + 4 dwords
#define A_BYTES  (RB * A_PITCH)           // 133,120
#define B_BYTES  (64u * B_PITCH)          // 66,560 per buffer
#define SMEM_BYTES (A_BYTES + 2u * B_BYTES)   // 266,240  (<= 320KB WGP LDS)

typedef __attribute__((ext_vector_type(16))) _Float16 v16h;
typedef __attribute__((ext_vector_type(8)))  _Float16 v8h;
typedef __attribute__((ext_vector_type(4)))  _Float16 v4h;
typedef __attribute__((ext_vector_type(8)))  float    v8f;
typedef __attribute__((ext_vector_type(4)))  unsigned v4u;
typedef __attribute__((ext_vector_type(8)))  int      v8i;
typedef __attribute__((ext_vector_type(4)))  int      v4i;

union Frag { v16h v; v8h h[2]; };

// ---------------- TDM: async DMA of one 64-row B tile (64 x 512 halfs) into LDS ----
// D# per CDNA5 ISA: 2D tensor, data_size=4B (dword elements),
// row = 256 dwords, LDS padding 4 dwords per 256 (pitch 1040B -> bank-conflict-free
// fragment reads), tile = 256 x 64.
__device__ __forceinline__ void tdm_load_b_tile(const _Float16* gsrc, unsigned ldsByteOff) {
#if __has_builtin(__builtin_amdgcn_tensor_load_to_lds)
  const unsigned long long ga = (unsigned long long)(const void*)gsrc;
  v4u g0;
  g0[0] = 1u;                                       // count=1 (valid), no gather
  g0[1] = ldsByteOff;                               // lds_addr (bytes)
  g0[2] = (unsigned)(ga & 0xFFFFFFFFull);           // global_addr[31:0]
  g0[3] = (unsigned)((ga >> 32) & 0x1FFFFFFull)     // global_addr[56:32]
        | (2u << 30);                               // type = 2 ("image")
  v8i g1;
  g1[0] = (2 << 16)                                 // data_size = 2 -> 4 bytes
        | (1 << 20)                                 // pad_enable
        | (7 << 22)                                 // pad_interval: every 256 dwords
        | (3 << 25);                                // pad_amount: 4 dwords
  g1[1] = (256 & 0xFFFF) << 16;                     // tensor_dim0 = 256 dwords/row
  g1[2] = (PPAD & 0xFFFF) << 16;                    // tensor_dim1 = PPAD rows
  g1[3] = (256 & 0xFFFF) << 16;                     // tile_dim0 = 256 dwords
  g1[4] = 64;                                       // tile_dim1 = 64 rows
  g1[5] = 256;                                      // tensor_dim0_stride = 256 dwords
  g1[6] = 0;
  g1[7] = 0;
  v4i gz = { 0, 0, 0, 0 };
#if defined(__clang_major__) && (__clang_major__ >= 23)
  v8i gz8 = { 0, 0, 0, 0, 0, 0, 0, 0 };
  __builtin_amdgcn_tensor_load_to_lds(g0, g1, gz, gz, gz8, 0);
#else
  __builtin_amdgcn_tensor_load_to_lds(g0, g1, gz, gz, 0);
#endif
#else
  (void)gsrc; (void)ldsByteOff;
#endif
}

__device__ __forceinline__ void tdm_wait_all() {
#if __has_builtin(__builtin_amdgcn_s_wait_tensorcnt)
  __builtin_amdgcn_s_wait_tensorcnt(0);
#endif
}

// ---------------- kernel 1: pair diff + normalize + diag ----------------
__global__ __launch_bounds__(128)
void pairs_norm_kernel(const float* __restrict__ txtf, const float* __restrict__ imgf,
                       _Float16* __restrict__ tdn, _Float16* __restrict__ imn,
                       float* __restrict__ diag) {
  const int p = blockIdx.x;
  const int t = threadIdx.x;

  if (p >= NPAIR) {             // zero the pad rows so GEMM pads are benign
    for (int d = t; d < DDIM; d += 128) {
      tdn[(size_t)p * DDIM + d] = (_Float16)0.0f;
      imn[(size_t)p * DDIM + d] = (_Float16)0.0f;
    }
    if (t == 0) diag[p] = 0.0f;
    return;
  }

  const int i = p / (NPTS - 1);
  const int r = p % (NPTS - 1);
  const int j = r + (r >= i ? 1 : 0);

  const float4* ti = (const float4*)(txtf + (size_t)i * DDIM);
  const float4* tj = (const float4*)(txtf + (size_t)j * DDIM);
  const float4* mi = (const float4*)(imgf + (size_t)i * DDIM);
  const float4* mj = (const float4*)(imgf + (size_t)j * DDIM);

  const float4 a = ti[t], b = tj[t], c = mi[t], d4 = mj[t];
  float td[4] = { a.x - b.x, a.y - b.y, a.z - b.z, a.w - b.w };
  float im[4] = { c.x - d4.x, c.y - d4.y, c.z - d4.z, c.w - d4.w };

  float ss_t = 0.f, ss_i = 0.f, dot = 0.f;
#pragma unroll
  for (int q = 0; q < 4; ++q) {
    ss_t += td[q] * td[q];
    ss_i += im[q] * im[q];
    dot  += td[q] * im[q];
  }
#pragma unroll
  for (int off = 16; off > 0; off >>= 1) {
    ss_t += __shfl_down(ss_t, off, 32);
    ss_i += __shfl_down(ss_i, off, 32);
    dot  += __shfl_down(dot , off, 32);
  }
  __shared__ float red[3][4];
  __shared__ float bc[2];
  const int wave = t >> 5, lane = t & 31;
  if (lane == 0) { red[0][wave] = ss_t; red[1][wave] = ss_i; red[2][wave] = dot; }
  __syncthreads();
  if (t == 0) {
    float st = red[0][0] + red[0][1] + red[0][2] + red[0][3];
    float si = red[1][0] + red[1][1] + red[1][2] + red[1][3];
    float dd = red[2][0] + red[2][1] + red[2][2] + red[2][3];
    float sc_t = 1.0f / (sqrtf(st) + EPSN);
    float sc_i = 1.0f / (sqrtf(si) + EPSN);
    bc[0] = sc_t; bc[1] = sc_i;
    diag[p] = INV_TEMP * dd * sc_t * sc_i;   // exact f32 diagonal of mma
  }
  __syncthreads();
  const float sc_t = bc[0], sc_i = bc[1];

  v4h ot, oi;
#pragma unroll
  for (int q = 0; q < 4; ++q) { ot[q] = (_Float16)(td[q] * sc_t); oi[q] = (_Float16)(im[q] * sc_i); }
  *(v4h*)(tdn + (size_t)p * DDIM + 4 * t) = ot;
  *(v4h*)(imn + (size_t)p * DDIM + 4 * t) = oi;
}

// ---------------- kernel 2: WMMA GEMM + fused online logsumexp ----------------
// C = X @ Y^T * 100 ; per-row (m, s) partials over this block's column split.
// Block: 256 threads = 8 waves; wave w owns rows [row0+16w, row0+16w+16).
// B tiles (64 rows x 512 halfs) arrive via TDM double-buffered async DMA.
__global__ __launch_bounds__(256)
void lse_pass_kernel(const _Float16* __restrict__ X, const _Float16* __restrict__ Y,
                     float* __restrict__ partial) {
  extern __shared__ __align__(16) unsigned char smem[];   // [A | B0 | B1]

  const int row0  = blockIdx.x * RB;
  const int split = blockIdx.y;
  const int t     = threadIdx.x;
  const int wave  = t >> 5, lane = t & 31, lo = lane & 15, hi = lane >> 4;
  const int ct0   = split * TPS;

  {   // cooperative stage of A: 128 rows x 512 halfs, pitch 1040B (conflict-free)
    const float4* gx = (const float4*)(X + (size_t)row0 * DDIM);
    for (int idx = t; idx < RB * 64; idx += 256) {     // 64 x 16B chunks per row
      const int rr = idx >> 6, cc = idx & 63;
      *(float4*)(smem + (size_t)rr * A_PITCH + (size_t)cc * 16) = gx[(size_t)rr * 64 + cc];
    }
  }

  // kick off the first B tile DMA (one TDM op per block, issued by wave 0)
  const unsigned ldsB0 = (unsigned)(unsigned long long)(const void*)(smem + A_BYTES);
  if (wave == 0) {
    tdm_load_b_tile(Y + (size_t)(ct0 * 64) * DDIM, ldsB0);
  }

  // A fragment source: row = wave*16 + lo (M = lane&15), K runs at hi*8 and hi*8+16
  const unsigned char* aRow = smem + (size_t)(wave * 16 + lo) * A_PITCH + hi * 16;

  const float NEG_INF = -__builtin_inff();
  float rm[8], rs[8];
#pragma unroll
  for (int q = 0; q < 8; ++q) { rm[q] = NEG_INF; rs[q] = 0.0f; }

  for (int it = 0; it < TPS; ++it) {
    const int ct = ct0 + it;

    if (wave == 0) tdm_wait_all();        // current buffer's DMA complete
    __syncthreads();                      // B[it&1] visible; B[(it+1)&1] free
    if (wave == 0 && it + 1 < TPS) {      // prefetch next tile while computing
      tdm_load_b_tile(Y + (size_t)((ct + 1) * 64) * DDIM,
                      ldsB0 + (unsigned)((it + 1) & 1) * B_BYTES);
    }

    const unsigned char* bbuf = smem + A_BYTES + (size_t)(it & 1) * B_BYTES;
    // B fragment source: lane lo = column within 16-wide subtile, K runs at hi*8 and hi*8+16
    const unsigned char* b0 = bbuf + (size_t)( 0 + lo) * B_PITCH + hi * 16;
    const unsigned char* b1 = bbuf + (size_t)(16 + lo) * B_PITCH + hi * 16;
    const unsigned char* b2 = bbuf + (size_t)(32 + lo) * B_PITCH + hi * 16;
    const unsigned char* b3 = bbuf + (size_t)(48 + lo) * B_PITCH + hi * 16;

    v8f acc0 = {}, acc1 = {}, acc2 = {}, acc3 = {};
#pragma unroll 4
    for (int k = 0; k < DDIM; k += 32) {
      Frag a;
      a.h[0] = *(const v8h*)(aRow + 2 * k);        // K = k + hi*8 + 0..7
      a.h[1] = *(const v8h*)(aRow + 2 * k + 32);   // K = k + hi*8 + 16..23
      Frag f0, f1, f2, f3;
      f0.h[0] = *(const v8h*)(b0 + 2 * k); f0.h[1] = *(const v8h*)(b0 + 2 * k + 32);
      f1.h[0] = *(const v8h*)(b1 + 2 * k); f1.h[1] = *(const v8h*)(b1 + 2 * k + 32);
      f2.h[0] = *(const v8h*)(b2 + 2 * k); f2.h[1] = *(const v8h*)(b2 + 2 * k + 32);
      f3.h[0] = *(const v8h*)(b3 + 2 * k); f3.h[1] = *(const v8h*)(b3 + 2 * k + 32);
      acc0 = __builtin_amdgcn_wmma_f32_16x16x32_f16(false, a.v, false, f0.v, (short)0, acc0, false, false);
      acc1 = __builtin_amdgcn_wmma_f32_16x16x32_f16(false, a.v, false, f1.v, (short)0, acc1, false, false);
      acc2 = __builtin_amdgcn_wmma_f32_16x16x32_f16(false, a.v, false, f2.v, (short)0, acc2, false, false);
      acc3 = __builtin_amdgcn_wmma_f32_16x16x32_f16(false, a.v, false, f3.v, (short)0, acc3, false, false);
    }

    // online logsumexp update: C layout -> lane holds col lo, rows hi*8 + q
    const int col_base = ct * 64;
#pragma unroll
    for (int sub = 0; sub < 4; ++sub) {
      const int gcol = col_base + sub * 16 + lo;
      if (gcol < NPAIR) {
        const v8f acc = (sub == 0) ? acc0 : (sub == 1) ? acc1 : (sub == 2) ? acc2 : acc3;
#pragma unroll
        for (int q = 0; q < 8; ++q) {
          const float v = acc[q] * INV_TEMP;
          if (v > rm[q]) { rs[q] = rs[q] * __expf(rm[q] - v) + 1.0f; rm[q] = v; }
          else           { rs[q] += __expf(v - rm[q]); }
        }
      }
    }
  }

  // merge (m,s) across the 16 lanes that share each row (vary in lo only)
#pragma unroll
  for (int off = 1; off < 16; off <<= 1) {
#pragma unroll
    for (int q = 0; q < 8; ++q) {
      const float om = __shfl_xor(rm[q], off, 32);
      const float os = __shfl_xor(rs[q], off, 32);
      const float nm = fmaxf(rm[q], om);
      rs[q] = rs[q] * __expf(rm[q] - nm) + os * __expf(om - nm);
      rm[q] = nm;
    }
  }

  if (lo == 0) {        // lanes 0 (rows hi*8=0..7) and 16 (rows 8..15) write
    const int growBase = row0 + wave * 16 + hi * 8;
#pragma unroll
    for (int q = 0; q < 8; ++q) {
      float* pp = partial + ((size_t)split * PPAD + growBase + q) * 2;
      pp[0] = rm[q];
      pp[1] = rs[q];
    }
  }
}

// ---------------- kernel 3: merge per-split partials into final LSE ----------------
__global__ __launch_bounds__(256)
void lse_merge_kernel(const float* __restrict__ partial, float* __restrict__ lse) {
  const int p = blockIdx.x * blockDim.x + threadIdx.x;
  if (p >= NPAIR) return;
  float m = -__builtin_inff();
#pragma unroll
  for (int s = 0; s < CSPLIT; ++s) m = fmaxf(m, partial[((size_t)s * PPAD + p) * 2]);
  float sum = 0.0f;
#pragma unroll
  for (int s = 0; s < CSPLIT; ++s) {
    const float mm = partial[((size_t)s * PPAD + p) * 2];
    const float ss = partial[((size_t)s * PPAD + p) * 2 + 1];
    sum += ss * __expf(mm - m);
  }
  lse[p] = m + __logf(sum);
}

// ---------------- kernel 4: final scalar loss ----------------
// loss = 0.5*(mean(rowLSE - diag) + mean(colLSE - diag))
__global__ __launch_bounds__(256)
void loss_kernel(const float* __restrict__ rowLSE, const float* __restrict__ colLSE,
                 const float* __restrict__ diag, float* __restrict__ out) {
  float acc = 0.0f;
  for (int p = threadIdx.x; p < NPAIR; p += 256)
    acc += rowLSE[p] + colLSE[p] - 2.0f * diag[p];
#pragma unroll
  for (int off = 16; off > 0; off >>= 1) acc += __shfl_down(acc, off, 32);
  __shared__ float red[8];
  const int wave = threadIdx.x >> 5;
  if ((threadIdx.x & 31) == 0) red[wave] = acc;
  __syncthreads();
  if (threadIdx.x == 0) {
    float tot = 0.0f;
#pragma unroll
    for (int w = 0; w < 8; ++w) tot += red[w];
    out[0] = tot / (2.0f * (float)NPAIR);
  }
}

// ---------------- workspace layout (bytes, 256-aligned) ----------------
#define OFF_TDN   ((size_t)0)
#define OFF_IMN   (OFF_TDN + (size_t)PPAD * DDIM * 2)            //  9,437,184
#define OFF_DIAG  (OFF_IMN + (size_t)PPAD * DDIM * 2)            // 18,874,368
#define OFF_ROW   (OFF_DIAG + (size_t)PPAD * 4)                  // 18,911,232
#define OFF_COL   (OFF_ROW + (size_t)PPAD * 4)                   // 18,948,096
#define OFF_PART  (OFF_COL + (size_t)PPAD * 4)                   // 18,984,960
// total: OFF_PART + CSPLIT*PPAD*2*4 = ~19.9 MB

extern "C" void kernel_launch(void* const* d_in, const int* in_sizes, int n_in,
                              void* d_out, int out_size, void* d_ws, size_t ws_size,
                              hipStream_t stream) {
  const float* txtf = (const float*)d_in[0];
  const float* imgf = (const float*)d_in[1];
  char* ws = (char*)d_ws;

  _Float16* tdn   = (_Float16*)(ws + OFF_TDN);
  _Float16* imn   = (_Float16*)(ws + OFF_IMN);
  float*    diag  = (float*)(ws + OFF_DIAG);
  float*    rowL  = (float*)(ws + OFF_ROW);
  float*    colL  = (float*)(ws + OFF_COL);
  float*    part  = (float*)(ws + OFF_PART);

  pairs_norm_kernel<<<dim3(PPAD), dim3(128), 0, stream>>>(txtf, imgf, tdn, imn, diag);

  // row logsumexp of  tdn @ imn^T * 100
  lse_pass_kernel<<<dim3(PPAD / RB, CSPLIT), dim3(256), SMEM_BYTES, stream>>>(tdn, imn, part);
  lse_merge_kernel<<<dim3((NPAIR + 255) / 256), dim3(256), 0, stream>>>(part, rowL);

  // col logsumexp == row logsumexp of the transpose: swap operands
  lse_pass_kernel<<<dim3(PPAD / RB, CSPLIT), dim3(256), SMEM_BYTES, stream>>>(imn, tdn, part);
  lse_merge_kernel<<<dim3((NPAIR + 255) / 256), dim3(256), 0, stream>>>(part, colL);

  loss_kernel<<<dim3(1), dim3(256), 0, stream>>>(rowL, colL, diag, (float*)d_out);
}
